// Attention4DDownsample_65180423685454
// MI455X (gfx1250) — matrix operand order repaired and checked by reading, not compiled, not run on hardware
//
#include <hip/hip_runtime.h>
#include <hip/hip_bf16.h>

// ---- vector types for WMMA fragments ----
typedef _Float16 h8x  __attribute__((ext_vector_type(8)));
typedef _Float16 h16x __attribute__((ext_vector_type(16)));
typedef float    f8x  __attribute__((ext_vector_type(8)));

union H16U { h16x v; h8x h[2]; };

#define DIMC   384
#define NHEAD  8
#define NHKD   128
#define DHC    512
#define OUTC   384
#define NPIX   49
#define SCALEF 0.25f      // KEY_DIM^-0.5
#define EPSB   1e-5f

// ---- d_ws layout ----
// [0, 13312)          cst: float2 {scale, shift} per channel
//                     [0,128) k   [128,256) q(incl attn scale)  [256,768) v
//                     [768,1280) v_local    [1280,1664) proj
// [16384, ...)        f16 weights (converted once per launch):
//                     kwh  @ half-offset 0       (128*384)
//                     qpwh @ half-offset 49152   (128*384)
//                     vwh  @ half-offset 98304   (512*384)
//                     pwh  @ half-offset 294912  (384*512)
// total = 16384 + 491520*2 = 999424 bytes
#define CST_K  0
#define CST_Q  128
#define CST_V  256
#define CST_VL 768
#define CST_P  1280
#define WH_OFF   16384
#define WH_KW    0
#define WH_QPW   49152
#define WH_VW    98304
#define WH_PW    294912
#define WH_TOTAL 491520

// LDS map (bytes), total 155648 (2 workgroups fit in a 320KB WGP):
//   [0      .. 49152)  xT   f16 [64][384]   (later: attn f32 [8][16][64] @0 ; outT f16 [16][512] @0)
//   [32768  .. 49152)  attnh f16 [8][16][64]
//   [49152  ..114688)  vh   f16 [512][64]   (temporarily qinT f16 [16][384] before v-GEMM)
//   [114688 ..147456)  khT  f16 [8][64][32] (kd 16..31 zero)
//   [147456 ..155648)  qTh  f16 [8][16][32] (kd 16..31 zero)

__device__ __forceinline__ h16x load_a_lds(const _Float16* base, int stride, int m0, int kbase, int lane) {
    int m  = lane & 15;
    int kb = (lane < 16) ? 0 : 8;
    const _Float16* p = base + (m0 + m) * stride + kbase + kb;
    H16U r;
    r.h[0] = *(const h8x*)(p);
    r.h[1] = *(const h8x*)(p + 16);
    return r.v;
}

__device__ __forceinline__ h16x load_b_lds(const _Float16* base, int stride, int n0, int kbase, int lane) {
    int n  = lane & 15;
    int ks = (lane < 16) ? 0 : 8;
    const _Float16* p = base + (n0 + n) * stride + kbase + ks;
    H16U r;
    r.h[0] = *(const h8x*)(p);
    r.h[1] = *(const h8x*)(p + 16);
    return r.v;
}

// A-fragment straight from pre-converted f16 weights in global (L2-resident)
__device__ __forceinline__ h16x load_a_gh(const _Float16* w, int ldw, int m0, int kbase, int lane) {
    int m  = lane & 15;
    int kb = (lane < 16) ? 0 : 8;
    const _Float16* p = w + (size_t)(m0 + m) * ldw + kbase + kb;
    H16U r;
    r.h[0] = *(const h8x*)(p);
    r.h[1] = *(const h8x*)(p + 16);
    return r.v;
}

// ---- prep kernel 1: fold BN (+conv bias, +attn scale) into per-channel {scale, shift} ----
__global__ __launch_bounds__(256)
void bnfold_kernel(const float* __restrict__ kb,  const float* __restrict__ kbn,
                   const float* __restrict__ qpb, const float* __restrict__ qbn,
                   const float* __restrict__ vb,  const float* __restrict__ vbn,
                   const float* __restrict__ vlb, const float* __restrict__ vlbn,
                   const float* __restrict__ pb,  const float* __restrict__ pbn,
                   float2* __restrict__ cst)
{
    int i = blockIdx.x * 256 + threadIdx.x;
    const float *bias, *bn;
    int c, C;
    float mul = 1.f;
    if (i < 128)       { bias = kb;  bn = kbn;  c = i;        C = 128; }
    else if (i < 256)  { bias = qpb; bn = qbn;  c = i - 128;  C = 128; mul = SCALEF; }
    else if (i < 768)  { bias = vb;  bn = vbn;  c = i - 256;  C = 512; }
    else if (i < 1280) { bias = vlb; bn = vlbn; c = i - 768;  C = 512; }
    else if (i < 1664) { bias = pb;  bn = pbn;  c = i - 1280; C = 384; }
    else return;
    float g = bn[c], be = bn[C + c], m = bn[2 * C + c], va = bn[3 * C + c];
    float s = g * rsqrtf(va + EPSB);
    float2 r;
    r.x = s * mul;
    r.y = ((bias[c] - m) * s + be) * mul;
    cst[i] = r;
}

// ---- prep kernel 2: convert all GEMM weights to f16 (row-major preserved) ----
__global__ __launch_bounds__(256)
void wcvt_kernel(const float* __restrict__ kw, const float* __restrict__ qpw,
                 const float* __restrict__ vw, const float* __restrict__ pw,
                 _Float16* __restrict__ wh)
{
    int i = blockIdx.x * 256 + threadIdx.x;
    if (i >= WH_TOTAL) return;
    float v;
    if (i < WH_QPW)      v = kw[i];
    else if (i < WH_VW)  v = qpw[i - WH_QPW];
    else if (i < WH_PW)  v = vw[i - WH_VW];
    else                 v = pw[i - WH_PW];
    wh[i] = (_Float16)v;
}

__global__ __launch_bounds__(256)
void attn4d_ds_kernel(const float* __restrict__ x,
                      const float* __restrict__ qlw, const float* __restrict__ qlb,
                      const _Float16* __restrict__ wh,
                      const float* __restrict__ vlw,
                      const float* __restrict__ ab,  const int* __restrict__ bidx, int n_off,
                      const float2* __restrict__ cst,
                      float* __restrict__ out)
{
    __shared__ __align__(16) unsigned char smem[155648];
    _Float16* xT    = (_Float16*)(smem);            // [64][384]
    float*    attn  = (float*)(smem);               // [8][16][64]
    _Float16* outT  = (_Float16*)(smem);            // [16][512]
    _Float16* attnh = (_Float16*)(smem + 32768);    // [8][16][64]
    _Float16* vh    = (_Float16*)(smem + 49152);    // [512][64]
    _Float16* qinT  = (_Float16*)(smem + 49152);    // [16][384]
    _Float16* khT   = (_Float16*)(smem + 114688);   // [8][64][32]
    _Float16* qTh   = (_Float16*)(smem + 147456);   // [8][16][32]

    const _Float16* kwh  = wh + WH_KW;
    const _Float16* qpwh = wh + WH_QPW;
    const _Float16* vwh  = wh + WH_VW;
    const _Float16* pwh  = wh + WH_PW;

    const int tid  = threadIdx.x;
    const int wave = tid >> 5;
    const int lane = tid & 31;
    const int b    = blockIdx.x;
    const float* xg = x + (size_t)b * (DIMC * NPIX);

    // warm L2 for proj weights (global_prefetch_b8)
    if (tid < 64) __builtin_prefetch(pwh + (size_t)tid * 3072, 0, 1);

    // ---- phase 1: stage x (transposed, f16) + zero pads ----
    for (int i = tid; i < 20480; i += 256) khT[i] = (_Float16)0.f;          // khT + qTh (contiguous)
    for (int i = tid; i < 15 * DIMC; i += 256) {                            // xT rows 49..63 = 0
        int n = 49 + i / DIMC;
        int c = i % DIMC;
        xT[n * DIMC + c] = (_Float16)0.f;
    }
    for (int i = tid; i < DIMC * NPIX; i += 256) {
        int c = i / NPIX, n = i % NPIX;
        xT[n * DIMC + c] = (_Float16)xg[i];
    }
    __syncthreads();

    // ---- phase 2: q input = depthwise3x3 s2 + avgpool2 (padded) ----
    for (int i = tid; i < DIMC * 16; i += 256) {
        int c = i >> 4, ql = i & 15, qy = ql >> 2, qx = ql & 3;
        float acc = qlb[c];
#pragma unroll
        for (int dy = 0; dy < 3; ++dy)
#pragma unroll
            for (int dx = 0; dx < 3; ++dx) {
                int iy = 2 * qy + dy - 1, ix = 2 * qx + dx - 1;
                if (iy >= 0 && iy < 7 && ix >= 0 && ix < 7)
                    acc += (float)xT[(iy * 7 + ix) * DIMC + c] * qlw[c * 9 + dy * 3 + dx];
            }
        if (qy < 3 && qx < 3) {
            float s = 0.f;
#pragma unroll
            for (int dy = 0; dy < 2; ++dy)
#pragma unroll
                for (int dx = 0; dx < 2; ++dx)
                    s += (float)xT[((2 * qy + dy) * 7 + (2 * qx + dx)) * DIMC + c];
            acc += 0.25f * s;
        }
        qinT[ql * DIMC + c] = (_Float16)acc;
    }
    __syncthreads();

    // ---- phase 3: k GEMM [128x384]x[384x64] (A reused over 4 n-tiles) + q GEMM ----
    {
        int m0 = wave * 16;
        f8x acc[4] = {};
#pragma unroll 1
        for (int kc = 0; kc < 12; ++kc) {
            h16x a = load_a_gh(kwh, DIMC, m0, kc * 32, lane);
#pragma unroll
            for (int nt = 0; nt < 4; ++nt) {
                h16x bm = load_b_lds(xT, DIMC, nt * 16, kc * 32, lane);
                acc[nt] = __builtin_amdgcn_wmma_f32_16x16x32_f16(false, a, false, bm, (short)0, acc[nt], false, false);
            }
        }
#pragma unroll
        for (int nt = 0; nt < 4; ++nt)
#pragma unroll
            for (int j = 0; j < 8; ++j) {
                int co = m0 + j + ((lane >= 16) ? 8 : 0);
                int ki = nt * 16 + (lane & 15);
                float2 c = cst[CST_K + co];
                float v = acc[nt][j] * c.x + c.y;
                khT[((co >> 4) * 64 + ki) * 32 + (co & 15)] = (_Float16)v;  // transposed, kd-padded
            }
    }
    {
        int m0 = wave * 16;
        f8x acc = {};
#pragma unroll 1
        for (int kc = 0; kc < 12; ++kc) {
            h16x a  = load_a_gh(qpwh, DIMC, m0, kc * 32, lane);
            h16x bm = load_b_lds(qinT, DIMC, 0, kc * 32, lane);
            acc = __builtin_amdgcn_wmma_f32_16x16x32_f16(false, a, false, bm, (short)0, acc, false, false);
        }
#pragma unroll
        for (int j = 0; j < 8; ++j) {
            int co = m0 + j + ((lane >= 16) ? 8 : 0);
            int qi = lane & 15;
            float2 c = cst[CST_Q + co];
            float v = acc[j] * c.x + c.y;     // BN + bias + attn scale, all folded
            qTh[((co >> 4) * 16 + qi) * 32 + (co & 15)] = (_Float16)v;
        }
    }
    __syncthreads();

    // ---- phase 4: v GEMM [512x384]x[384x64], A reused over 4 n-tiles ----
#pragma unroll 1
    for (int mi = 0; mi < 4; ++mi) {
        int m0 = (wave + mi * 8) * 16;
        f8x acc[4] = {};
#pragma unroll 1
        for (int kc = 0; kc < 12; ++kc) {
            h16x a = load_a_gh(vwh, DIMC, m0, kc * 32, lane);
#pragma unroll
            for (int nt = 0; nt < 4; ++nt) {
                h16x bm = load_b_lds(xT, DIMC, nt * 16, kc * 32, lane);
                acc[nt] = __builtin_amdgcn_wmma_f32_16x16x32_f16(false, a, false, bm, (short)0, acc[nt], false, false);
            }
        }
#pragma unroll
        for (int nt = 0; nt < 4; ++nt)
#pragma unroll
            for (int j = 0; j < 8; ++j) {
                int d = m0 + j + ((lane >= 16) ? 8 : 0);
                int n = nt * 16 + (lane & 15);
                float2 c = cst[CST_V + d];
                float v = acc[nt][j] * c.x + c.y;
                vh[d * 64 + n] = (_Float16)v;
            }
    }
    __syncthreads();

    // ---- phase 5: attn = (q^T k) per head, K=32 (kd padded) -> f32 over dead xT ----
#pragma unroll 1
    for (int nt = 0; nt < 4; ++nt) {
        int h = wave, n0 = nt * 16;
        f8x acc = {};
        h16x a  = load_a_lds(qTh + h * 16 * 32, 32, 0, 0, lane);
        h16x bm = load_b_lds(khT + h * 64 * 32, 32, n0, 0, lane);
        acc = __builtin_amdgcn_wmma_f32_16x16x32_f16(false, a, false, bm, (short)0, acc, false, false);
#pragma unroll
        for (int j = 0; j < 8; ++j) {
            int qi = j + ((lane >= 16) ? 8 : 0);
            int ki = n0 + (lane & 15);
            attn[(h * 16 + qi) * 64 + ki] = acc[j];
        }
    }
    __syncthreads();

    // ---- phase 6: +bias, softmax over 49, -> f16 (k 49..63 zero) ----
    if (tid < 128) {
        int h = tid >> 4, qi = tid & 15;
        float* row = attn + (h * 16 + qi) * 64;
        const int* bi = bidx + qi * NPIX;
        const float* abh = ab + (size_t)h * n_off;
        float mx = -1e30f;
        for (int k = 0; k < NPIX; ++k) {
            float v = row[k] + abh[bi[k]];
            row[k] = v;
            mx = fmaxf(mx, v);
        }
        float s = 0.f;
        for (int k = 0; k < NPIX; ++k) {
            float e = __expf(row[k] - mx);
            row[k] = e;
            s += e;
        }
        float inv = 1.f / s;
        _Float16* hrow = attnh + (h * 16 + qi) * 64;
        for (int k = 0; k < NPIX; ++k) hrow[k] = (_Float16)(row[k] * inv);
        for (int k = NPIX; k < 64; ++k) hrow[k] = (_Float16)0.f;
    }
    __syncthreads();

    // ---- phase 7: xa = v @ attn^T  (+ v_local depthwise + ReLU) -> outT f16 ----
#pragma unroll 1
    for (int dt = 0; dt < 4; ++dt) {
        int h  = wave;
        int d0 = h * 64 + dt * 16;
        f8x acc = {};
#pragma unroll 1
        for (int kc = 0; kc < 2; ++kc) {
            h16x a  = load_a_lds(vh, 64, d0, kc * 32, lane);
            h16x bm = load_b_lds(attnh + h * 16 * 64, 64, 0, kc * 32, lane);
            acc = __builtin_amdgcn_wmma_f32_16x16x32_f16(false, a, false, bm, (short)0, acc, false, false);
        }
#pragma unroll
        for (int j = 0; j < 8; ++j) {
            int d  = d0 + j + ((lane >= 16) ? 8 : 0);
            int qi = lane & 15, qy = qi >> 2, qx = qi & 3;
            float t = 0.f;
#pragma unroll
            for (int dy = 0; dy < 3; ++dy)
#pragma unroll
                for (int dx = 0; dx < 3; ++dx) {
                    int iy = 2 * qy + dy - 1, ix = 2 * qx + dx - 1;
                    if (iy >= 0 && iy < 7 && ix >= 0 && ix < 7)
                        t += (float)vh[d * 64 + iy * 7 + ix] * vlw[d * 9 + dy * 3 + dx];
                }
            float2 c = cst[CST_VL + d];
            float vl = t * c.x + c.y;        // depthwise bias + BN folded
            float v = fmaxf(acc[j] + vl, 0.f);
            outT[qi * 512 + d] = (_Float16)v;   // over dead attn-f32 region
        }
    }
    __syncthreads();

    // ---- phase 8: proj GEMM [384x512]x[512x16] + BN -> global ----
#pragma unroll 1
    for (int i2 = 0; i2 < 3; ++i2) {
        int m0 = (wave * 3 + i2) * 16;
        f8x acc = {};
#pragma unroll 1
        for (int kc = 0; kc < 16; ++kc) {
            h16x a  = load_a_gh(pwh, DHC, m0, kc * 32, lane);
            h16x bm = load_b_lds(outT, DHC, 0, kc * 32, lane);
            acc = __builtin_amdgcn_wmma_f32_16x16x32_f16(false, a, false, bm, (short)0, acc, false, false);
        }
#pragma unroll
        for (int j = 0; j < 8; ++j) {
            int co = m0 + j + ((lane >= 16) ? 8 : 0);
            int qi = lane & 15;
            float2 c = cst[CST_P + co];
            float v = acc[j] * c.x + c.y;
            out[(size_t)b * (OUTC * 16) + co * 16 + qi] = v;
        }
    }
}

extern "C" void kernel_launch(void* const* d_in, const int* in_sizes, int n_in,
                              void* d_out, int out_size, void* d_ws, size_t ws_size,
                              hipStream_t stream) {
    (void)n_in; (void)out_size; (void)ws_size;
    const float* x    = (const float*)d_in[0];
    const float* qlw  = (const float*)d_in[1];
    const float* qlb  = (const float*)d_in[2];
    const float* qpw  = (const float*)d_in[3];
    const float* qpb  = (const float*)d_in[4];
    const float* qbn  = (const float*)d_in[5];
    const float* kw   = (const float*)d_in[6];
    const float* kb   = (const float*)d_in[7];
    const float* kbn  = (const float*)d_in[8];
    const float* vw   = (const float*)d_in[9];
    const float* vb   = (const float*)d_in[10];
    const float* vbn  = (const float*)d_in[11];
    const float* vlw  = (const float*)d_in[12];
    const float* vlb  = (const float*)d_in[13];
    const float* vlbn = (const float*)d_in[14];
    const float* pw   = (const float*)d_in[15];
    const float* pb   = (const float*)d_in[16];
    const float* pbn  = (const float*)d_in[17];
    const float* ab   = (const float*)d_in[18];
    const int*   bidx = (const int*)d_in[19];

    int n_off = in_sizes[18] / NHEAD;
    int B     = in_sizes[0] / (DIMC * NPIX);

    float2*    cst = (float2*)d_ws;
    _Float16*  wh  = (_Float16*)((char*)d_ws + WH_OFF);

    bnfold_kernel<<<7, 256, 0, stream>>>(kb, kbn, qpb, qbn, vb, vbn, vlb, vlbn, pb, pbn, cst);
    wcvt_kernel<<<(WH_TOTAL + 255) / 256, 256, 0, stream>>>(kw, qpw, vw, pw, wh);
    attn4d_ds_kernel<<<B, 256, 0, stream>>>(x, qlw, qlb, wh, vlw,
                                            ab, bidx, n_off, cst, (float*)d_out);
}
